// CombinedGraphLayer_8778913153237
// MI455X (gfx1250) — matrix operand order repaired and checked by reading, not compiled, not run on hardware
//
#include <hip/hip_runtime.h>
#include <hip/hip_bf16.h>

typedef __attribute__((ext_vector_type(16))) __bf16 bf16x16;
typedef __attribute__((ext_vector_type(8)))  float  f32x8;

#define DEVFN __device__ __forceinline__

constexpr int       kB    = 16;
constexpr int       kN    = 12800;
constexpr int       kF    = 256;
constexpr int       kD    = 128;
constexpr int       kNB   = 100;   // bins per batch
constexpr int       kBS   = 128;   // bin size
constexpr int       kOut  = 256;
constexpr long long kRows = (long long)kB * kN;   // 204800
constexpr int       kKeys = 199;   // bin_idx in [0,198]

// Padded LDS leading dims: addr/4 stride == 4 (mod 16) -> 16 lanes x 4-bank b128
// accesses tile all 64 banks, conflict-free.
constexpr int kLdX  = 264;  // for 256-wide row-major A tiles
constexpr int kLdH  = 136;  // for 128-deep column-major / 128-wide row-major
constexpr int kLdA  = 72;   // for 64-deep tiles
constexpr int kLdW  = 40;   // for 32-deep weight tiles
constexpr int kWSZ  = 128 * kLdW;   // one weight-tile buffer
constexpr int kASZ  = 128 * kLdA;   // one FFN tile buffer

// ---------------------------------------------------------------------------
// WMMA helpers (CDNA5 v_wmma_f32_16x16x32_bf16, wave32)
// ---------------------------------------------------------------------------
DEVFN f32x8 zero8() {
  f32x8 z;
#pragma unroll
  for (int i = 0; i < 8; ++i) z[i] = 0.f;
  return z;
}

DEVFN f32x8 wmma_bf16(bf16x16 a, bf16x16 b, f32x8 c) {
  return __builtin_amdgcn_wmma_f32_16x16x32_bf16(false, a, false, b, (short)0, c,
                                                 false, false);
}

// A fragment (16x32 bf16) from row-major [row][k] LDS; two b128 loads per lane.
DEVFN bf16x16 frag_a(const __bf16* s, int ld, int row0, int k0, int lane) {
  int half = (lane >> 4) & 1, r = lane & 15;
  const __bf16* p = s + (size_t)(row0 + r) * ld + k0 + half * 8;
  bf16x16 a;
  *(uint4*)&a       = *(const uint4*)p;          // a[0..7]
  *((uint4*)&a + 1) = *(const uint4*)(p + 16);   // a[8..15]
  return a;
}

// B fragment (32x16 bf16) from column-major [n][k] LDS; two b128 loads per lane.
DEVFN bf16x16 frag_bc(const __bf16* s, int ldk, int k0, int col0, int lane) {
  int half = (lane >> 4) & 1, c = lane & 15;
  const __bf16* p = s + (size_t)(col0 + c) * ldk + k0 + half * 8;
  bf16x16 b;
  *(uint4*)&b       = *(const uint4*)p;          // b[0..7]
  *((uint4*)&b + 1) = *(const uint4*)(p + 16);   // b[8..15]
  return b;
}

// Fold x*mskf (mskf in {0,1}) into the A fragment: vector select, no cvt chains
DEVFN bf16x16 mask_frag(bf16x16 a, float m) {
  bf16x16 z;
#pragma unroll
  for (int i = 0; i < 16; ++i) z[i] = (__bf16)0.f;
  return (m != 0.f) ? a : z;
}

// 1-deep software-pipelined 8-tile WMMA chain: the loads of tile t+1 are issued
// before the WMMA of tile t, so LDS latency overlaps matrix execution with only
// two live B fragments.
DEVFN void wmma_row8(const bf16x16& af, const __bf16* s, int ldk, int k0,
                     int colbase, int lane, f32x8* acc) {
  bf16x16 bcur = frag_bc(s, ldk, k0, colbase, lane);
#pragma unroll
  for (int t = 0; t < 7; ++t) {
    bf16x16 bnxt = frag_bc(s, ldk, k0, colbase + 16 * (t + 1), lane);
    acc[t] = wmma_bf16(af, bcur, acc[t]);
    bcur = bnxt;
  }
  acc[7] = wmma_bf16(af, bcur, acc[7]);
}

DEVFN float elu1(float x) { return x > 0.f ? x : expm1f(x); }

union Pack8 { uint4 q; __bf16 h[8]; };

// Cooperative copy: [128n][32k] weight tile (N-major global) -> col-major LDS
DEVFN void coop_w32(__bf16* dst, const __bf16* wT, int nbase, int k0, int tid) {
  for (int i = tid; i < 128 * 16; i += 256) {
    int n = i >> 4, cu = i & 15;
    ((unsigned*)dst)[n * (kLdW / 2) + cu] =
        ((const unsigned*)(wT + (size_t)(nbase + n) * 256 + k0))[cu];
  }
}

// ---------------------------------------------------------------------------
// f32 -> bf16 convert with transpose: in[k*N+n] -> out[n*K+k]  (N-major weights)
// ---------------------------------------------------------------------------
__global__ void k_cvt_t(const float* __restrict__ in, __bf16* __restrict__ out,
                        int K, int N) {
  int total = K * N;
  for (int i = blockIdx.x * blockDim.x + threadIdx.x; i < total;
       i += gridDim.x * blockDim.x) {
    int k = i / N, n = i - k * N;
    out[n * K + k] = (__bf16)in[i];
  }
}

// ---------------------------------------------------------------------------
// LayerNorm: one wave per 256-wide row -> bf16
// ---------------------------------------------------------------------------
__global__ void __launch_bounds__(256) k_layernorm(const float* __restrict__ x,
                                                   const float* __restrict__ g,
                                                   const float* __restrict__ be,
                                                   __bf16* __restrict__ out) {
  int lane = threadIdx.x & 31, w = threadIdx.x >> 5;
  long long row = (long long)blockIdx.x * 8 + w;
  const float4* xr = (const float4*)(x + row * kF);
  float4 a = xr[lane * 2], b = xr[lane * 2 + 1];
  float v[8] = {a.x, a.y, a.z, a.w, b.x, b.y, b.z, b.w};
  float s = 0.f;
#pragma unroll
  for (int i = 0; i < 8; ++i) s += v[i];
#pragma unroll
  for (int off = 16; off > 0; off >>= 1) s += __shfl_xor(s, off, 32);
  float mu = s * (1.f / kF);
  float q = 0.f;
#pragma unroll
  for (int i = 0; i < 8; ++i) { float d = v[i] - mu; q += d * d; }
#pragma unroll
  for (int off = 16; off > 0; off >>= 1) q += __shfl_xor(q, off, 32);
  float rs = rsqrtf(q * (1.f / kF) + 1e-6f);
  Pack8 u;
#pragma unroll
  for (int i = 0; i < 8; ++i) {
    int col = lane * 8 + i;
    u.h[i] = (__bf16)((v[i] - mu) * rs * g[col] + be[col]);
  }
  *(uint4*)(out + row * kF + lane * 8) = u.q;
}

// ---------------------------------------------------------------------------
// Fused FFN: xdist = elu(xn@W1 + b1)@W2 + b2, 128-row tile per block,
// double-buffered K tiles, pipelined B fragments.
// ---------------------------------------------------------------------------
__global__ void __launch_bounds__(256) k_ffn(const __bf16* __restrict__ xn,
                                             const __bf16* __restrict__ w1t,
                                             const float* __restrict__ b1,
                                             const __bf16* __restrict__ w2t,
                                             const float* __restrict__ b2,
                                             __bf16* __restrict__ xd) {
  __shared__ __bf16 sA[2 * kASZ];       // activation tiles [128][64] pad 72
  __shared__ __bf16 sW[2 * kASZ];       // W1 tiles col-major [128n][64k]
  __shared__ __bf16 sH[128 * kLdH];     // hidden row-major [128][128] pad 136
  __shared__ __bf16 sW2[128 * kLdH];    // W2 col-major [128n][128k] pad 136
  __shared__ float  sB1[128], sB2[128];
  int tid = threadIdx.x, lane = tid & 31, w = tid >> 5;
  long long row0 = (long long)blockIdx.x * 128;

  for (int i = tid; i < 128 * 64; i += 256) {
    int n = i >> 6, cu = i & 63;
    ((unsigned*)sW2)[n * (kLdH / 2) + cu] = ((const unsigned*)w2t)[n * 64 + cu];
  }
  if (tid < 128) { sB1[tid] = b1[tid]; sB2[tid] = b2[tid]; }

  // prime tile 0
  for (int i = tid; i < 128 * 32; i += 256) {
    int r = i >> 5, cu = i & 31;
    ((unsigned*)sA)[r * (kLdA / 2) + cu] =
        ((const unsigned*)(xn + (row0 + r) * kF))[cu];
    ((unsigned*)sW)[r * (kLdA / 2) + cu] = ((const unsigned*)(w1t + (size_t)r * kF))[cu];
  }

  f32x8 acc[8];
#pragma unroll
  for (int t = 0; t < 8; ++t) acc[t] = zero8();

  for (int ks = 0; ks < 4; ++ks) {
    __syncthreads();
    int cur = ks & 1;
    if (ks < 3) {   // prefetch next K tile into the other buffer
      int k0 = (ks + 1) * 64;
      for (int i = tid; i < 128 * 32; i += 256) {
        int r = i >> 5, cu = i & 31;
        ((unsigned*)(sA + (cur ^ 1) * kASZ))[r * (kLdA / 2) + cu] =
            ((const unsigned*)(xn + (row0 + r) * kF + k0))[cu];
        ((unsigned*)(sW + (cur ^ 1) * kASZ))[r * (kLdA / 2) + cu] =
            ((const unsigned*)(w1t + (size_t)r * kF + k0))[cu];
      }
    }
#pragma unroll
    for (int kk = 0; kk < 64; kk += 32) {
      bf16x16 af = frag_a(sA + cur * kASZ, kLdA, 16 * w, kk, lane);
      wmma_row8(af, sW + cur * kASZ, kLdA, kk, 0, lane, acc);
    }
  }
  __syncthreads();
  int half = (lane >> 4) & 1, cn = lane & 15;
  int rowbase = 16 * w + 8 * half;
#pragma unroll
  for (int t = 0; t < 8; ++t)
#pragma unroll
    for (int r = 0; r < 8; ++r) {
      int col = 16 * t + cn;
      sH[(rowbase + r) * kLdH + col] = (__bf16)elu1(acc[t][r] + sB1[col]);
    }
  __syncthreads();
  f32x8 acc2[8];
#pragma unroll
  for (int t = 0; t < 8; ++t) acc2[t] = zero8();
#pragma unroll
  for (int kk = 0; kk < 128; kk += 32) {
    bf16x16 af = frag_a(sH, kLdH, 16 * w, kk, lane);
    wmma_row8(af, sW2, kLdH, kk, 0, lane, acc2);
  }
#pragma unroll
  for (int t = 0; t < 8; ++t)
#pragma unroll
    for (int r = 0; r < 8; ++r) {
      int col = 16 * t + cn;
      xd[(row0 + rowbase + r) * kD + col] = (__bf16)(acc2[t][r] + sB2[col]);
    }
}

// ---------------------------------------------------------------------------
// LSH bin index: argmax over [xdist@cb, -xdist@cb] (+99 for masked points)
// ---------------------------------------------------------------------------
__global__ void __launch_bounds__(256) k_lsh(const __bf16* __restrict__ xd,
                                             const float* __restrict__ cb,
                                             const unsigned char* __restrict__ msk,
                                             int* __restrict__ bidx) {
  __shared__ float sx[8][128];
  int lane = threadIdx.x & 31, w = threadIdx.x >> 5;
  long long p = (long long)blockIdx.x * 8 + w;
  const __bf16* xr = xd + p * kD;
#pragma unroll
  for (int i = 0; i < 4; ++i) sx[w][lane * 4 + i] = (float)xr[lane * 4 + i];
  __syncthreads();
  float bv = -1e30f;
  int bi = 0x7fffffff;
  for (int c = lane; c < 50; c += 32) {
    float dot = 0.f;
    for (int k = 0; k < 128; ++k) dot += sx[w][k] * cb[k * kNB + c];
    if (dot > bv || (dot == bv && c < bi)) { bv = dot; bi = c; }
    float nd = -dot;
    int ci = c + 50;
    if (nd > bv || (nd == bv && ci < bi)) { bv = nd; bi = ci; }
  }
#pragma unroll
  for (int off = 16; off > 0; off >>= 1) {
    float ov = __shfl_xor(bv, off, 32);
    int   oi = __shfl_xor(bi, off, 32);
    if (ov > bv || (ov == bv && oi < bi)) { bv = ov; bi = oi; }
  }
  if (lane == 0) bidx[p] = bi + (msk[p] ? 0 : (kNB - 1));
}

// ---------------------------------------------------------------------------
// Stable counting sort (== argsort of bin keys) per batch.
// ---------------------------------------------------------------------------
__global__ void __launch_bounds__(128) k_sort(const int* __restrict__ bidx,
                                              int* __restrict__ perm) {
  __shared__ int cnt[128 * kKeys];
  __shared__ int base[kKeys];
  __shared__ int tot[kKeys];
  int t = threadIdx.x;
  int b = blockIdx.x;
  const int* keys = bidx + (long long)b * kN;
  for (int i = t; i < 128 * kKeys; i += 128) cnt[i] = 0;
  __syncthreads();
  for (int e = 0; e < 100; ++e) cnt[t * kKeys + keys[t * 100 + e]]++;
  __syncthreads();
  if (t < kKeys) {
    int run = 0;
    for (int c = 0; c < 128; ++c) {
      int idx = c * kKeys + t;
      int v = cnt[idx];
      cnt[idx] = run;
      run += v;
    }
    tot[t] = run;
  }
  __syncthreads();
  if (t == 0) {
    int acc = 0;
    for (int j = 0; j < kKeys; ++j) { base[j] = acc; acc += tot[j]; }
  }
  __syncthreads();
  for (int e = 0; e < 100; ++e) {
    int k = keys[t * 100 + e];
    int o = cnt[t * kKeys + k];
    cnt[t * kKeys + k] = o + 1;
    perm[(long long)b * kN + base[k] + o] = t * 100 + e;
  }
}

// ---------------------------------------------------------------------------
// Row gathers (bf16 rows moved as u32) + mask gather
// ---------------------------------------------------------------------------
__global__ void k_gather_rows(const unsigned* __restrict__ in,
                              unsigned* __restrict__ out,
                              const int* __restrict__ perm, int wu) {
  long long row = blockIdx.x;
  int b = (int)(row / kN);
  long long src = (long long)b * kN + perm[row];
  for (int t = threadIdx.x; t < wu; t += blockDim.x)
    out[row * wu + t] = in[src * wu + t];
}

__global__ void k_gather_mask(const unsigned char* __restrict__ msk,
                              const int* __restrict__ perm,
                              float* __restrict__ mb) {
  long long i = (long long)blockIdx.x * blockDim.x + threadIdx.x;
  if (i >= kRows) return;
  int b = (int)(i / kN);
  mb[i] = msk[(long long)b * kN + perm[i]] ? 1.f : 0.f;
}

// ---------------------------------------------------------------------------
// Per-bin Gaussian kernel: gram via WMMA (X@X^T), dm (symmetric, transposed
// b128 stores), degree -> norm
// ---------------------------------------------------------------------------
__global__ void __launch_bounds__(256) k_gram(const __bf16* __restrict__ xdb,
                                              const float* __restrict__ mb,
                                              __bf16* __restrict__ dm,
                                              float* __restrict__ normb) {
  __shared__ __bf16 sXd[128 * kLdH];
  __shared__ float  sna[128], sm[128], sdeg[128];
  int tid = threadIdx.x, lane = tid & 31, w = tid >> 5;
  long long bin = blockIdx.x;
  const unsigned* src = (const unsigned*)(xdb + bin * (128LL * 128));
  for (int i = tid; i < 128 * 64; i += 256) {
    int r = i >> 6, cu = i & 63;
    ((unsigned*)sXd)[r * (kLdH / 2) + cu] = src[i];
  }
  if (tid < 128) { sm[tid] = mb[bin * 128 + tid]; sdeg[tid] = 0.f; }
  __syncthreads();
  if (tid < 128) {
    float s = 0.f;
    for (int k = 0; k < 128; ++k) { float v = (float)sXd[tid * kLdH + k]; s += v * v; }
    sna[tid] = s;
  }
  __syncthreads();
  f32x8 acc[8];
#pragma unroll
  for (int t = 0; t < 8; ++t) acc[t] = zero8();
#pragma unroll
  for (int kk = 0; kk < 128; kk += 32) {
    bf16x16 af = frag_a(sXd, kLdH, 16 * w, kk, lane);
    wmma_row8(af, sXd, kLdH, kk, 0, lane, acc);  // B[k][n]=Xd[n][k]: col-major
  }
  int half = (lane >> 4) & 1, cn = lane & 15;
  int rowbase = 16 * w + 8 * half;
  float rsum[8];
#pragma unroll
  for (int r = 0; r < 8; ++r) rsum[r] = 0.f;
#pragma unroll
  for (int t = 0; t < 8; ++t) {
    int col = 16 * t + cn;
    Pack8 u;
#pragma unroll
    for (int r = 0; r < 8; ++r) {
      int row = rowbase + r;
      float d2 = sna[row] - 2.f * acc[t][r] + sna[col];
      float dist = sqrtf(fmaxf(d2, 1e-6f));
      float v = fminf(__expf(-0.1f * dist), 1.f);   // CLIP_LOW=0 implicit
      v *= sm[row] * sm[col];
      rsum[r] += v;
      u.h[r] = (__bf16)v;
    }
    // dm is symmetric: store tile transposed so each lane writes 16B contiguous
    *(uint4*)(dm + bin * 16384 + (size_t)col * 128 + rowbase) = u.q;
  }
#pragma unroll
  for (int r = 0; r < 8; ++r) atomicAdd(&sdeg[rowbase + r], rsum[r]);
  __syncthreads();
  if (tid < 128) {
    float dg = fminf(fmaxf(sdeg[tid], 0.f), 1000.f);
    normb[bin * 128 + tid] = rsqrtf(dg + 1e-6f) * sm[tid];
  }
}

// ---------------------------------------------------------------------------
// GHConv layer: one workgroup per bin, all GEMMs out of LDS (~285KB of 320KB),
// double-buffered weight tiles, pipelined B fragments.
// ---------------------------------------------------------------------------
template <int FINAL>
__global__ void __launch_bounds__(256) k_ghconv(
    const __bf16* __restrict__ xin, const __bf16* __restrict__ dm,
    const float* __restrict__ normb, const float* __restrict__ mb,
    const int* __restrict__ perm, const __bf16* __restrict__ WtT,
    const float* __restrict__ bt, const __bf16* __restrict__ WhT,
    const __bf16* __restrict__ ThT, __bf16* __restrict__ out_bf,
    float* __restrict__ out_f32) {
  __shared__ __bf16 sX[128 * kLdX];     // X row-major [128][256] pad 264 (66KB)
  __shared__ __bf16 sPT[256 * kLdH];    // P  col-major [256n][128k] pad 136 (68KB)
  __shared__ __bf16 sHT[256 * kLdH];    // f_hom col-major (68KB)
  __shared__ __bf16 sAdj[128 * kLdH];   // adj row-major [128][128] pad 136 (34KB)
  __shared__ __bf16 sWa[2 * kWSZ];      // double-buffered weight tiles (20KB)
  __shared__ __bf16 sWb[2 * kWSZ];      // (20KB)
  __shared__ float  sNorm[128], sM[128], sBt[256];
  int tid = threadIdx.x, lane = tid & 31, w = tid >> 5;
  long long bin = blockIdx.x;
  int half = (lane >> 4) & 1, cn = lane & 15;
  int rowbase = 16 * w + 8 * half;

  // Warm L2 for this layer's weights (global_prefetch_b8)
#pragma unroll
  for (int it = 0; it < 4; ++it) {
    __builtin_prefetch((const char*)WtT + (size_t)it * 32768 + tid * 128, 0, 1);
    __builtin_prefetch((const char*)WhT + (size_t)it * 32768 + tid * 128, 0, 1);
    __builtin_prefetch((const char*)ThT + (size_t)it * 32768 + tid * 128, 0, 1);
  }

  const unsigned* xs = (const unsigned*)(xin + bin * (128LL * 256));
  for (int i = tid; i < 128 * 128; i += 256) {
    int r = i >> 7, cu = i & 127;
    ((unsigned*)sX)[r * (kLdX / 2) + cu] = xs[i];
  }
  const unsigned* as = (const unsigned*)(dm + bin * (128LL * 128));
  for (int i = tid; i < 128 * 64; i += 256) {
    int r = i >> 6, cu = i & 63;
    ((unsigned*)sAdj)[r * (kLdH / 2) + cu] = as[i];
  }
  if (tid < 128) { sNorm[tid] = normb[bin * 128 + tid]; sM[tid] = mb[bin * 128 + tid]; }
  sBt[tid] = bt[tid];
  coop_w32(sWa, ThT, 0, 0, tid);   // prime P-phase tile (ch=0, k0=0)
  __syncthreads();
  float myM = sM[16 * w + (lane & 15)];   // mask for this lane's A-fragment row

  // ---- P = (X*mskf)@Th * norm  (staged column-major into sPT) ----
  for (int ch = 0; ch < 2; ++ch) {
    f32x8 acc[8];
#pragma unroll
    for (int t = 0; t < 8; ++t) acc[t] = zero8();
    for (int ks = 0; ks < 8; ++ks) {
      if (ch || ks) __syncthreads();
      int cur = ks & 1;
      if (ks < 7) coop_w32(sWa + (cur ^ 1) * kWSZ, ThT, ch * 128, (ks + 1) * 32, tid);
      else if (ch == 0) coop_w32(sWa + (cur ^ 1) * kWSZ, ThT, 128, 0, tid);
      bf16x16 am = mask_frag(frag_a(sX, kLdX, 16 * w, ks * 32, lane), myM);
      wmma_row8(am, sWa + cur * kWSZ, kLdW, 0, 0, lane, acc);
    }
#pragma unroll
    for (int t = 0; t < 8; ++t) {
      int col = ch * 128 + 16 * t + cn;
      Pack8 u;
#pragma unroll
      for (int r = 0; r < 8; ++r)
        u.h[r] = (__bf16)(acc[t][r] * sNorm[rowbase + r]);
      *(uint4*)(sPT + (size_t)col * kLdH + rowbase) = u.q;
    }
  }
  __syncthreads();

  // ---- f_hom = (adj @ P) * norm  (staged column-major into sHT) ----
  for (int ch = 0; ch < 2; ++ch) {
    f32x8 acc[8];
#pragma unroll
    for (int t = 0; t < 8; ++t) acc[t] = zero8();
#pragma unroll
    for (int k0 = 0; k0 < 128; k0 += 32) {
      bf16x16 aa = frag_a(sAdj, kLdH, 16 * w, k0, lane);
      wmma_row8(aa, sPT, kLdH, k0, ch * 128, lane, acc);
    }
#pragma unroll
    for (int t = 0; t < 8; ++t) {
      int col = ch * 128 + 16 * t + cn;
      Pack8 u;
#pragma unroll
      for (int r = 0; r < 8; ++r)
        u.h[r] = (__bf16)(acc[t][r] * sNorm[rowbase + r]);
      *(uint4*)(sHT + (size_t)col * kLdH + rowbase) = u.q;
    }
  }
  __syncthreads();
  coop_w32(sWa, WtT, 0, 0, tid);   // prime gate/het tiles (ch=0, k0=0)
  coop_w32(sWb, WhT, 0, 0, tid);

  // ---- gate & f_het, then combine ----
  for (int ch = 0; ch < 2; ++ch) {
    f32x8 aG[8], aH[8];
#pragma unroll
    for (int t = 0; t < 8; ++t) { aG[t] = zero8(); aH[t] = zero8(); }
    for (int ks = 0; ks < 8; ++ks) {
      __syncthreads();
      int cur = ks & 1;
      if (ks < 7) {
        coop_w32(sWa + (cur ^ 1) * kWSZ, WtT, ch * 128, (ks + 1) * 32, tid);
        coop_w32(sWb + (cur ^ 1) * kWSZ, WhT, ch * 128, (ks + 1) * 32, tid);
      } else if (ch == 0) {
        coop_w32(sWa + (cur ^ 1) * kWSZ, WtT, 128, 0, tid);
        coop_w32(sWb + (cur ^ 1) * kWSZ, WhT, 128, 0, tid);
      }
      bf16x16 ax = frag_a(sX, kLdX, 16 * w, ks * 32, lane);
      bf16x16 am = mask_frag(ax, myM);
      // two interleaved 1-deep pipelines (gate via Wt, het via Wh)
      bf16x16 bg = frag_bc(sWa + cur * kWSZ, kLdW, 0, 0, lane);
      bf16x16 bh = frag_bc(sWb + cur * kWSZ, kLdW, 0, 0, lane);
#pragma unroll
      for (int t = 0; t < 7; ++t) {
        bf16x16 bgn = frag_bc(sWa + cur * kWSZ, kLdW, 0, 16 * (t + 1), lane);
        bf16x16 bhn = frag_bc(sWb + cur * kWSZ, kLdW, 0, 16 * (t + 1), lane);
        aG[t] = wmma_bf16(ax, bg, aG[t]);
        aH[t] = wmma_bf16(am, bh, aH[t]);
        bg = bgn;
        bh = bhn;
      }
      aG[7] = wmma_bf16(ax, bg, aG[7]);
      aH[7] = wmma_bf16(am, bh, aH[7]);
    }
#pragma unroll
    for (int t = 0; t < 8; ++t) {
      int col = ch * 128 + 16 * t + cn;
      Pack8 u;
      u.q = *(const uint4*)(sHT + (size_t)col * kLdH + rowbase);
#pragma unroll
      for (int r = 0; r < 8; ++r) {
        int row = rowbase + r;
        float g = 1.f / (1.f + __expf(-(aG[t][r] + sBt[col])));
        float o = g * (float)u.h[r] + (1.f - g) * aH[t][r];
        o = elu1(o) * sM[row];
        if (FINAL) {
          int b = (int)(bin / kNB);
          int pos = (int)(bin % kNB) * 128 + row;
          int orig = perm[(long long)b * kN + pos];
          out_f32[((long long)b * kN + orig) * kOut + col] = o;
        } else {
          out_bf[(bin * 128 + row) * kOut + col] = (__bf16)o;
        }
      }
    }
  }
}

// ---------------------------------------------------------------------------
extern "C" void kernel_launch(void* const* d_in, const int* in_sizes, int n_in,
                              void* d_out, int out_size, void* d_ws, size_t ws_size,
                              hipStream_t stream) {
  (void)in_sizes; (void)n_in; (void)out_size; (void)ws_size;
  const float*         x   = (const float*)d_in[0];
  const unsigned char* msk = (const unsigned char*)d_in[1];
  const float* lng = (const float*)d_in[3];
  const float* lnb = (const float*)d_in[4];
  const float* W1  = (const float*)d_in[5];
  const float* b1  = (const float*)d_in[6];
  const float* W2  = (const float*)d_in[7];
  const float* b2  = (const float*)d_in[8];
  const float* cb  = (const float*)d_in[9];
  const float* Wt0 = (const float*)d_in[10];
  const float* bt0 = (const float*)d_in[11];
  const float* Wh0 = (const float*)d_in[12];
  const float* th0 = (const float*)d_in[13];
  const float* Wt1 = (const float*)d_in[14];
  const float* bt1 = (const float*)d_in[15];
  const float* Wh1 = (const float*)d_in[16];
  const float* th1 = (const float*)d_in[17];

  char* ws = (char*)d_ws;
  size_t off = 0;
  auto alloc = [&](size_t bytes) -> char* {
    char* p = ws + off;
    off += (bytes + 255) & ~(size_t)255;
    return p;
  };
  __bf16* xnbf  = (__bf16*)alloc((size_t)kRows * kF * 2);
  __bf16* xdbf  = (__bf16*)alloc((size_t)kRows * kD * 2);
  int*    bidx  = (int*)   alloc((size_t)kRows * 4);
  int*    perm  = (int*)   alloc((size_t)kRows * 4);
  __bf16* xb    = (__bf16*)alloc((size_t)kRows * kF * 2);
  __bf16* xdb   = (__bf16*)alloc((size_t)kRows * kD * 2);
  float*  mb    = (float*) alloc((size_t)kRows * 4);
  __bf16* dmb   = (__bf16*)alloc((size_t)kB * kNB * kBS * kBS * 2);
  float*  normb = (float*) alloc((size_t)kRows * 4);
  __bf16* h0    = (__bf16*)alloc((size_t)kRows * kOut * 2);
  __bf16* w1t   = (__bf16*)alloc((size_t)kF * kD * 2);
  __bf16* w2t   = (__bf16*)alloc((size_t)kD * kD * 2);
  __bf16* wt0t  = (__bf16*)alloc((size_t)kF * kOut * 2);
  __bf16* wh0t  = (__bf16*)alloc((size_t)kF * kOut * 2);
  __bf16* th0t  = (__bf16*)alloc((size_t)kF * kOut * 2);
  __bf16* wt1t  = (__bf16*)alloc((size_t)kOut * kOut * 2);
  __bf16* wh1t  = (__bf16*)alloc((size_t)kOut * kOut * 2);
  __bf16* th1t  = (__bf16*)alloc((size_t)kOut * kOut * 2);

  k_cvt_t<<<64, 256, 0, stream>>>(W1, w1t, kF, kD);
  k_cvt_t<<<64, 256, 0, stream>>>(W2, w2t, kD, kD);
  k_cvt_t<<<256, 256, 0, stream>>>(Wt0, wt0t, kF, kOut);
  k_cvt_t<<<256, 256, 0, stream>>>(Wh0, wh0t, kF, kOut);
  k_cvt_t<<<256, 256, 0, stream>>>(th0, th0t, kF, kOut);
  k_cvt_t<<<256, 256, 0, stream>>>(Wt1, wt1t, kOut, kOut);
  k_cvt_t<<<256, 256, 0, stream>>>(Wh1, wh1t, kOut, kOut);
  k_cvt_t<<<256, 256, 0, stream>>>(th1, th1t, kOut, kOut);

  k_layernorm<<<(unsigned)(kRows / 8), 256, 0, stream>>>(x, lng, lnb, xnbf);
  k_ffn<<<(unsigned)(kRows / 128), 256, 0, stream>>>(xnbf, w1t, b1, w2t, b2, xdbf);
  k_lsh<<<(unsigned)(kRows / 8), 256, 0, stream>>>(xdbf, cb, msk, bidx);
  k_sort<<<kB, 128, 0, stream>>>(bidx, perm);
  k_gather_rows<<<(unsigned)kRows, 128, 0, stream>>>((const unsigned*)xnbf,
                                                     (unsigned*)xb, perm, kF / 2);
  k_gather_rows<<<(unsigned)kRows, 128, 0, stream>>>((const unsigned*)xdbf,
                                                     (unsigned*)xdb, perm, kD / 2);
  k_gather_mask<<<(unsigned)((kRows + 255) / 256), 256, 0, stream>>>(msk, perm, mb);
  k_gram<<<kB * kNB, 256, 0, stream>>>(xdb, mb, dmb, normb);
  k_ghconv<0><<<kB * kNB, 256, 0, stream>>>(xb, dmb, normb, mb, perm, wt0t, bt0,
                                            wh0t, th0t, h0, nullptr);
  k_ghconv<1><<<kB * kNB, 256, 0, stream>>>(h0, dmb, normb, mb, perm, wt1t, bt1,
                                            wh1t, th1t, nullptr, (float*)d_out);
}
